// Model_27307402068683
// MI455X (gfx1250) — matrix operand order repaired and checked by reading, not compiled, not run on hardware
//
#include <hip/hip_runtime.h>
#include <hip/hip_bf16.h>
#include <math.h>

// ---------------- problem constants (match reference) ----------------
#define NN      50000
#define MPAD    50016          // NN rounded up to 32 (for 32-row wave tiles)
#define EE      800000
#define F_IN    128
#define HDIM    256
#define BB      128
#define NCLS    10
#define SCL     3
#define NNZV    1200000

typedef __attribute__((ext_vector_type(16))) __bf16 v16bf;
typedef __attribute__((ext_vector_type(8)))  __bf16 v8bf;
typedef __attribute__((ext_vector_type(8)))  float  v8f;

// ---------------- elementwise / setup kernels ----------------
__global__ void k_fill_f32(float* p, long n, float v) {
    long i = (long)blockIdx.x * blockDim.x + threadIdx.x;
    if (i < n) p[i] = v;
}

__global__ void k_deg_scatter(const int* __restrict__ ei, float* __restrict__ deg) {
    long e = (long)blockIdx.x * blockDim.x + threadIdx.x;
    if (e < EE) atomicAdd(&deg[ei[EE + e]], 1.0f);   // dst row
}

__global__ void k_rsqrt_inplace(float* p, long n) {
    long i = (long)blockIdx.x * blockDim.x + threadIdx.x;
    if (i < n) p[i] = rsqrtf(p[i]);                  // deg >= 1 guaranteed (self loops)
}

// converts n_valid elements; zero-fills up to n_total (padded GEMM rows)
__global__ void k_f32_to_bf16(const float* __restrict__ in, __bf16* __restrict__ out,
                              long n_valid, long n_total, int relu) {
    long i = (long)blockIdx.x * blockDim.x + threadIdx.x;
    if (i < n_total) {
        float v = (i < n_valid) ? in[i] : 0.0f;
        if (relu) v = fmaxf(v, 0.0f);
        out[i] = (__bf16)v;
    }
}

// W [K,Nc] f32 row-major  ->  Wt [Nc,K] bf16 (so B fragments load contiguously)
__global__ void k_w_to_bf16_t(const float* __restrict__ W, __bf16* __restrict__ Wt,
                              int K, int Nc) {
    int i = blockIdx.x * blockDim.x + threadIdx.x;
    if (i < K * Nc) {
        int k = i / Nc, n = i % Nc;
        Wt[(size_t)n * K + k] = (__bf16)W[i];
    }
}

__global__ void k_relu_inplace(float* p, long n) {
    long i = (long)blockIdx.x * blockDim.x + threadIdx.x;
    if (i < n) p[i] = fmaxf(p[i], 0.0f);
}

// ---------------- WMMA bf16 GEMM: C[M,Nc] = A[M,K] * Wt[Nc,K]^T ----------------
// register-blocked: one wave32 computes a 32x64 C tile (2 M-tiles x 4 N-tiles,
// 8 f32 accumulators). Per 32-deep K step: 4 A b128 + 8 B b128 loads feed
// 8 v_wmma_f32_16x16x32_bf16 (1.5 loads per WMMA; A reused 4x, B reused 2x).
__global__ __launch_bounds__(128)
void k_gemm_bf16(const __bf16* __restrict__ A, const __bf16* __restrict__ Bt,
                 float* __restrict__ C, int M, int K, int Nc) {
    int wave   = blockIdx.x * (blockDim.x >> 5) + (threadIdx.x >> 5);
    int tilesN = Nc >> 6;                      // 64-wide N groups
    int total  = (M >> 5) * tilesN;            // 32-tall M groups
    if (wave >= total) return;                 // wave-uniform: EXEC stays all-ones
    int tm2 = wave / tilesN;                   // 32-row group
    int tn4 = wave - tm2 * tilesN;             // 64-col group

    int lane = threadIdx.x & 31;
    int half = lane >> 4;                      // lane group (0: lanes 0-15, 1: 16-31)
    int l15  = lane & 15;

    // A fragments: rows m = tm2*32 + i*16 + l15
    //   element idx -> k = kk + 16*half + idx
    const __bf16* arow0 = A + (size_t)(tm2 * 32 + l15) * K + half * 16;
    const __bf16* arow1 = arow0 + (size_t)16 * K;
    // B fragments: cols n = tn4*64 + j*16 + l15; element idx -> k = kk + 16*half + idx
    const __bf16* bbase = Bt + (size_t)(tn4 * 64 + l15) * K + half * 16;

    v8f acc[2][4];
#pragma unroll
    for (int i = 0; i < 2; ++i)
#pragma unroll
        for (int j = 0; j < 4; ++j) acc[i][j] = (v8f){};

    for (int kk = 0; kk < K; kk += 32) {
        v16bf a[2];
        {
            v8bf lo0 = *(const v8bf*)(arow0 + kk);
            v8bf hi0 = *(const v8bf*)(arow0 + kk + 8);
            a[0] = __builtin_shufflevector(lo0, hi0,
                       0,1,2,3,4,5,6,7,8,9,10,11,12,13,14,15);
            v8bf lo1 = *(const v8bf*)(arow1 + kk);
            v8bf hi1 = *(const v8bf*)(arow1 + kk + 8);
            a[1] = __builtin_shufflevector(lo1, hi1,
                       0,1,2,3,4,5,6,7,8,9,10,11,12,13,14,15);
        }
        v16bf b[4];
#pragma unroll
        for (int j = 0; j < 4; ++j)
            b[j] = *(const v16bf*)(bbase + (size_t)j * 16 * K + kk);
#pragma unroll
        for (int i = 0; i < 2; ++i)
#pragma unroll
            for (int j = 0; j < 4; ++j)
                acc[i][j] = __builtin_amdgcn_wmma_f32_16x16x32_bf16(
                                false, a[i], false, b[j], (short)0, acc[i][j],
                                false, false);
    }

    // C/D layout: VGPR r holds M = r + 8*half, N = l15
#pragma unroll
    for (int i = 0; i < 2; ++i) {
#pragma unroll
        for (int j = 0; j < 4; ++j) {
            float* crow = C + (size_t)(tm2 * 32 + i * 16 + half * 8) * Nc
                            + tn4 * 64 + j * 16 + l15;
#pragma unroll
            for (int r = 0; r < 8; ++r) crow[(size_t)r * Nc] = acc[i][j][r];
        }
    }
}

// ---------------- GCN aggregation ----------------
// agg = dinv[i]^2 * mbuf[i] + bias   (self-loop term + bias folded in; no atomics)
__global__ void k_init_agg(const float* __restrict__ mbuf, const float* __restrict__ dinv,
                           const float* __restrict__ bias, float* __restrict__ agg) {
    long i = (long)blockIdx.x * blockDim.x + threadIdx.x;
    if (i < (long)NN * HDIM) {
        int node = (int)(i / HDIM);
        int f    = (int)(i - (long)node * HDIM);
        float dv = dinv[node];
        agg[i] = dv * dv * mbuf[i] + bias[f];
    }
}

// agg[dst] += dinv[src]*dinv[dst] * mbuf[src]   — one thread per (edge, 4 feats)
__global__ void k_edge_scatter(const int* __restrict__ ei, const float* __restrict__ dinv,
                               const float* __restrict__ mbuf, float* __restrict__ agg) {
    long t = (long)blockIdx.x * blockDim.x + threadIdx.x;
    const int Q = HDIM / 4;
    if (t >= (long)EE * Q) return;
    int e = (int)(t / Q);
    int f = (int)(t - (long)e * Q) * 4;
    int s = ei[e], dd = ei[EE + e];
    float w = dinv[s] * dinv[dd];
    const float4 v = *(const float4*)(mbuf + (size_t)s * HDIM + f);
    float* o = agg + (size_t)dd * HDIM + f;
    atomicAdd(o + 0, w * v.x);
    atomicAdd(o + 1, w * v.y);
    atomicAdd(o + 2, w * v.z);
    atomicAdd(o + 3, w * v.w);
}

// ---------------- fused wavelet transform + per-(scale,graph) pooling ----------------
// pooled[scale*B + batch[r%N]][f] += d[nz] * h3[col][f]   (skips 153MB xt buffer)
__global__ void k_wavelet_pool(const int* __restrict__ didx, const float* __restrict__ dval,
                               const int* __restrict__ batch, const float* __restrict__ h3,
                               float* __restrict__ pooled) {
    long t = (long)blockIdx.x * blockDim.x + threadIdx.x;
    const int Q = HDIM / 4;
    if (t >= (long)NNZV * Q) return;
    long nz = t / Q;
    int f = (int)(t - nz * Q) * 4;
    int r = didx[nz];
    int c = didx[NNZV + nz];
    int scale = r / NN;
    int node  = r - scale * NN;
    int g = batch[node];
    float w = dval[nz];
    const float4 v = *(const float4*)(h3 + (size_t)c * HDIM + f);
    float* o = pooled + ((size_t)(scale * BB + g)) * HDIM + f;
    atomicAdd(o + 0, w * v.x);
    atomicAdd(o + 1, w * v.y);
    atomicAdd(o + 2, w * v.z);
    atomicAdd(o + 3, w * v.w);
}

// ---------------- FC head (tiny; VALU is fine) ----------------
// pooled stored [SCALE*B, H]; logical fc1 input index k = s*H + hh maps to pooled[s*B+b][hh]
__global__ void k_fc1(const float* __restrict__ pooled, const float* __restrict__ W,
                      const float* __restrict__ bv, float* __restrict__ z1) {
    int t = blockIdx.x * blockDim.x + threadIdx.x;
    if (t >= BB * HDIM) return;
    int b = t / HDIM, j = t - b * HDIM;
    float acc = bv[j];
    for (int s = 0; s < SCL; ++s) {
        const float* p = pooled + ((size_t)(s * BB + b)) * HDIM;
        const float* w = W + ((size_t)s * HDIM) * HDIM + j;
        for (int hh = 0; hh < HDIM; ++hh) acc += p[hh] * w[(size_t)hh * HDIM];
    }
    z1[t] = fmaxf(acc, 0.0f);
}

__global__ void k_fc2(const float* __restrict__ z1, const float* __restrict__ W,
                      const float* __restrict__ bv, float* __restrict__ z2) {
    const int H2 = HDIM / 2;
    int t = blockIdx.x * blockDim.x + threadIdx.x;
    if (t >= BB * H2) return;
    int b = t / H2, j = t - b * H2;
    float acc = bv[j];
    const float* p = z1 + (size_t)b * HDIM;
    for (int k = 0; k < HDIM; ++k) acc += p[k] * W[(size_t)k * H2 + j];
    z2[t] = fmaxf(acc, 0.0f);
}

// one wave32 per batch row; shuffle reductions (wave32-safe)
__global__ void k_fc3_logsoftmax(const float* __restrict__ z2, const float* __restrict__ W,
                                 const float* __restrict__ bv, float* __restrict__ out) {
    const int H2 = HDIM / 2;
    int b = blockIdx.x;
    int lane = threadIdx.x;                 // 0..31
    int c = lane < NCLS ? lane : NCLS - 1;  // clamp to stay in-bounds
    float acc = bv[c];
    const float* p = z2 + (size_t)b * H2;
    for (int k = 0; k < H2; ++k) acc += p[k] * W[(size_t)k * NCLS + c];
    float v = (lane < NCLS) ? acc : -1e30f;
    float mx = v;
    for (int o = 16; o; o >>= 1) mx = fmaxf(mx, __shfl_xor(mx, o, 32));
    float ex = (lane < NCLS) ? expf(v - mx) : 0.0f;
    float sum = ex;
    for (int o = 16; o; o >>= 1) sum += __shfl_xor(sum, o, 32);
    if (lane < NCLS) out[b * NCLS + lane] = v - mx - logf(sum);
}

// ---------------- host launch ----------------
static inline int nblk(long n, int b) { return (int)((n + b - 1) / b); }

extern "C" void kernel_launch(void* const* d_in, const int* in_sizes, int n_in,
                              void* d_out, int out_size, void* d_ws, size_t ws_size,
                              hipStream_t stream) {
    const float* x     = (const float*)d_in[0];
    const float* W1    = (const float*)d_in[1];
    const float* b1    = (const float*)d_in[2];
    const float* W2    = (const float*)d_in[3];
    const float* b2    = (const float*)d_in[4];
    const float* W3    = (const float*)d_in[5];
    const float* b3    = (const float*)d_in[6];
    const float* fc1w  = (const float*)d_in[7];
    const float* fc1b  = (const float*)d_in[8];
    const float* fc2w  = (const float*)d_in[9];
    const float* fc2b  = (const float*)d_in[10];
    const float* fc3w  = (const float*)d_in[11];
    const float* fc3b  = (const float*)d_in[12];
    const float* dval  = (const float*)d_in[13];
    const int*   ei    = (const int*)d_in[14];
    const int*   batch = (const int*)d_in[15];
    const int*   didx  = (const int*)d_in[16];
    float* out = (float*)d_out;

    // workspace layout (256B aligned slices)
    char* ws = (char*)d_ws;
    size_t off = 0;
    auto alloc = [&](size_t bytes) -> void* {
        void* p = ws + off;
        off = (off + bytes + 255) & ~(size_t)255;
        return p;
    };
    float*  dinv   = (float*) alloc((size_t)NN * 4);
    __bf16* h_bf   = (__bf16*)alloc((size_t)MPAD * HDIM * 2);
    __bf16* Wt_bf  = (__bf16*)alloc((size_t)HDIM * HDIM * 2);
    float*  mbuf   = (float*) alloc((size_t)MPAD * HDIM * 4);
    float*  agg    = (float*) alloc((size_t)NN * HDIM * 4);
    float*  pooled = (float*) alloc((size_t)SCL * BB * HDIM * 4);
    float*  z1     = (float*) alloc((size_t)BB * HDIM * 4);
    float*  z2     = (float*) alloc((size_t)BB * (HDIM / 2) * 4);

    const int T = 256;

    // ---- degree -> dinv (self loop gives deg >= 1) ----
    k_fill_f32<<<nblk(NN, T), T, 0, stream>>>(dinv, NN, 1.0f);
    k_deg_scatter<<<nblk(EE, T), T, 0, stream>>>(ei, dinv);
    k_rsqrt_inplace<<<nblk(NN, T), T, 0, stream>>>(dinv, NN);

    long nh   = (long)NN * HDIM;
    long nedf = (long)EE * (HDIM / 4);
    int  gemm_waves = (MPAD / 32) * (HDIM / 64);   // 32x64 C tile per wave
    int  gemm_blks  = nblk(gemm_waves, 4);         // 4 waves / 128-thread block

    // ---- GCN layer 1 (K = F_IN) ----
    k_f32_to_bf16<<<nblk((long)MPAD * F_IN, T), T, 0, stream>>>(
        x, h_bf, (long)NN * F_IN, (long)MPAD * F_IN, 0);
    k_w_to_bf16_t<<<nblk(F_IN * HDIM, T), T, 0, stream>>>(W1, Wt_bf, F_IN, HDIM);
    k_gemm_bf16<<<gemm_blks, 128, 0, stream>>>(h_bf, Wt_bf, mbuf, MPAD, F_IN, HDIM);
    k_init_agg<<<nblk(nh, T), T, 0, stream>>>(mbuf, dinv, b1, agg);
    k_edge_scatter<<<nblk(nedf, T), T, 0, stream>>>(ei, dinv, mbuf, agg);

    // ---- GCN layer 2 (ReLU fused into bf16 convert) ----
    k_f32_to_bf16<<<nblk((long)MPAD * HDIM, T), T, 0, stream>>>(
        agg, h_bf, nh, (long)MPAD * HDIM, 1);
    k_w_to_bf16_t<<<nblk(HDIM * HDIM, T), T, 0, stream>>>(W2, Wt_bf, HDIM, HDIM);
    k_gemm_bf16<<<gemm_blks, 128, 0, stream>>>(h_bf, Wt_bf, mbuf, MPAD, HDIM, HDIM);
    k_init_agg<<<nblk(nh, T), T, 0, stream>>>(mbuf, dinv, b2, agg);
    k_edge_scatter<<<nblk(nedf, T), T, 0, stream>>>(ei, dinv, mbuf, agg);

    // ---- GCN layer 3 ----
    k_f32_to_bf16<<<nblk((long)MPAD * HDIM, T), T, 0, stream>>>(
        agg, h_bf, nh, (long)MPAD * HDIM, 1);
    k_w_to_bf16_t<<<nblk(HDIM * HDIM, T), T, 0, stream>>>(W3, Wt_bf, HDIM, HDIM);
    k_gemm_bf16<<<gemm_blks, 128, 0, stream>>>(h_bf, Wt_bf, mbuf, MPAD, HDIM, HDIM);
    k_init_agg<<<nblk(nh, T), T, 0, stream>>>(mbuf, dinv, b3, agg);
    k_edge_scatter<<<nblk(nedf, T), T, 0, stream>>>(ei, dinv, mbuf, agg);
    k_relu_inplace<<<nblk(nh, T), T, 0, stream>>>(agg, nh);   // h3 (f32) for wavelet

    // ---- fused wavelet transform + pooling ----
    long npool = (long)SCL * BB * HDIM;
    k_fill_f32<<<nblk(npool, T), T, 0, stream>>>(pooled, npool, 0.0f);
    k_wavelet_pool<<<nblk((long)NNZV * (HDIM / 4), T), T, 0, stream>>>(
        didx, dval, batch, agg, pooled);

    // ---- FC head + log_softmax ----
    k_fc1<<<nblk(BB * HDIM, T), T, 0, stream>>>(pooled, fc1w, fc1b, z1);
    k_fc2<<<nblk(BB * (HDIM / 2), T), T, 0, stream>>>(z1, fc2w, fc2b, z2);
    k_fc3_logsoftmax<<<BB, 32, 0, stream>>>(z2, fc3w, fc3b, out);
}
